// RTDetrMultiscaleDeformableAttention_69114613730725
// MI455X (gfx1250) — matrix operand order repaired and checked by reading, not compiled, not run on hardware
//
#include <hip/hip_runtime.h>
#include <hip/hip_bf16.h>

typedef __attribute__((ext_vector_type(16))) __bf16 v16bf;
typedef __attribute__((ext_vector_type(8)))  float  v8f;

// ---------- helpers ----------
__device__ __forceinline__ unsigned int pack_bf16_trunc(float lo, float hi) {
    // one v_perm_b32: {hi.b3, hi.b2, lo.b3, lo.b2}
    return __builtin_amdgcn_perm(__float_as_uint(hi), __float_as_uint(lo), 0x07060302u);
}

__device__ __forceinline__ unsigned int pack_bf16_rne(float lo, float hi) {
    unsigned int ulo = __float_as_uint(lo);
    unsigned int uhi = __float_as_uint(hi);
    ulo += 0x7fffu + ((ulo >> 16) & 1u);
    uhi += 0x7fffu + ((uhi >> 16) & 1u);
    return (ulo >> 16) | (uhi & 0xffff0000u);
}

__device__ __forceinline__ unsigned short f32_to_bf16_rne(float f) {
    unsigned int u = __float_as_uint(f);
    u += 0x7fffu + ((u >> 16) & 1u);
    return (unsigned short)(u >> 16);
}

__device__ __forceinline__ float bf16_bits_to_f32(unsigned short s) {
    return __uint_as_float(((unsigned int)s) << 16);
}

// LDS byte offset (relative to workgroup LDS base) of a __shared__ object.
__device__ __forceinline__ unsigned int lds_offset_of(const void* p) {
    return (unsigned int)(uintptr_t)(__attribute__((address_space(3))) const void*)p;
}

// ---------- 1. swizzle 256x256 f32 weight into bf16 B-fragment layout ----------
// Layout: [ntile(16)][kstep(8)][lane(32)][j(8)] uint32 (2 packed bf16).
// B fragment (16x16x32 bf16): lanes 0-15 hold K=0..7,16..23 (V_j -> K pairs),
// lanes 16-31 hold K=8..15,24..31. col = lane & 15.
__global__ void swizzle_w_kernel(const float* __restrict__ W, unsigned int* __restrict__ out) {
    int tid   = blockIdx.x * blockDim.x + threadIdx.x;   // 0..32767
    int j     = tid & 7;
    int lane  = (tid >> 3) & 31;
    int kstep = (tid >> 8) & 7;
    int ntile = tid >> 11;
    int col   = lane & 15;
    int khalf = lane >> 4;
    int k = kstep * 32 + (j >> 2) * 16 + khalf * 8 + 2 * (j & 3);
    int n = ntile * 16 + col;
    out[tid] = pack_bf16_rne(W[k * 256 + n], W[(k + 1) * 256 + n]);   // one-time: keep RNE
}

// ---------- 2. WMMA GEMM: C(M x 256) = A(M x 256, f32) * Bsw + bias ----------
// Block = 8 waves sharing one n-group (64 cols). The block's 32 KB B slice is
// staged LDS-resident once via async-to-LDS, then each wave computes a 16x64
// strip over 8 k-steps: 4 x v_wmma_f32_16x16x32_bf16 per k-step, B from LDS.
union AFrag { unsigned int u[8]; v16bf v; };
union BFrag { uint4 q[2];        v16bf v; };

__device__ __forceinline__ void load_a_frag(const float* __restrict__ ap, AFrag& f) {
    float4 a0 = *(const float4*)(ap);         // K = kbase+0..3
    float4 a1 = *(const float4*)(ap + 4);     // K = kbase+4..7
    float4 a2 = *(const float4*)(ap + 16);    // K = kbase+16..19
    float4 a3 = *(const float4*)(ap + 20);    // K = kbase+20..23
    f.u[0] = pack_bf16_trunc(a0.x, a0.y); f.u[1] = pack_bf16_trunc(a0.z, a0.w);
    f.u[2] = pack_bf16_trunc(a1.x, a1.y); f.u[3] = pack_bf16_trunc(a1.z, a1.w);
    f.u[4] = pack_bf16_trunc(a2.x, a2.y); f.u[5] = pack_bf16_trunc(a2.z, a2.w);
    f.u[6] = pack_bf16_trunc(a3.x, a3.y); f.u[7] = pack_bf16_trunc(a3.z, a3.w);
}

template <bool BF16_OUT>
__global__ __launch_bounds__(256)
void gemm_bf16_wmma_kernel(const float* __restrict__ A,
                           const unsigned int* __restrict__ Bsw,
                           const float* __restrict__ bias,
                           void* __restrict__ Cout, int Mtiles) {
    __shared__ __align__(16) unsigned int lds_b[8192];   // 32 KB B slice
    int tid  = threadIdx.x;
    int wid  = tid >> 5;
    int lane = tid & 31;
    int ng     = blockIdx.x & 3;                  // n-group: cols [ng*64, ng*64+64)
    int mgroup = blockIdx.x >> 2;
    int mtile  = mgroup * 8 + wid;

    // ---- stage this n-group's 32 KB swizzled-B slice into LDS (async DMA) ----
    {
        const unsigned int* gsrc = Bsw + (size_t)ng * 8192;   // contiguous slice
        unsigned int lofs = lds_offset_of(&lds_b[tid * 4]);   // 16 B per thread per chunk
        unsigned int voff = (unsigned int)(tid * 16);
        // INST_OFFSET is added to BOTH the LDS and global addresses.
#pragma unroll
        for (int c = 0; c < 8; ++c)
            asm volatile("global_load_async_to_lds_b128 %0, %1, %2 offset:%c3"
                         :: "v"(lofs), "v"(voff), "s"(gsrc), "i"(c * 4096)
                         : "memory");
        asm volatile("s_wait_asynccnt 0x0" ::: "memory");
    }
    __syncthreads();
    if (mtile >= Mtiles) return;                  // wave-uniform; EXEC stays all-1

    int row = mtile * 16 + (lane & 15);           // A: lane%16 = M row
    const float* Abase = A + (size_t)row * 256 + (lane >> 4) * 8;   // + khalf*8
    const unsigned int* lbase = &lds_b[lane * 8]; // + (ks*32)*8 per k-step, +t*2048 per tile

    v8f c0 = {0.f,0.f,0.f,0.f,0.f,0.f,0.f,0.f};
    v8f c1 = c0, c2 = c0, c3 = c0;

    AFrag af;
    load_a_frag(Abase, af);

#pragma unroll
    for (int ks = 0; ks < 8; ++ks) {
        AFrag afn;
        if (ks < 7)                               // software-pipeline A 1 k-step ahead
            load_a_frag(Abase + (ks + 1) * 32, afn);
        if (ks + 2 < 8)                           // A-stream prefetch (global_prefetch_b8)
            __builtin_prefetch(Abase + (ks + 2) * 32, 0, 3);

        const unsigned int* bk = lbase + ks * 256;
        BFrag b0, b1, b2, b3;                     // ds_load_b128 x8 (LDS-resident B)
        b0.q[0] = *(const uint4*)(bk);            b0.q[1] = *(const uint4*)(bk + 4);
        b1.q[0] = *(const uint4*)(bk + 2048);     b1.q[1] = *(const uint4*)(bk + 2052);
        b2.q[0] = *(const uint4*)(bk + 4096);     b2.q[1] = *(const uint4*)(bk + 4100);
        b3.q[0] = *(const uint4*)(bk + 6144);     b3.q[1] = *(const uint4*)(bk + 6148);

        c0 = __builtin_amdgcn_wmma_f32_16x16x32_bf16(false, af.v, false, b0.v, (short)0, c0, false, false);
        c1 = __builtin_amdgcn_wmma_f32_16x16x32_bf16(false, af.v, false, b1.v, (short)0, c1, false, false);
        c2 = __builtin_amdgcn_wmma_f32_16x16x32_bf16(false, af.v, false, b2.v, (short)0, c2, false, false);
        c3 = __builtin_amdgcn_wmma_f32_16x16x32_bf16(false, af.v, false, b3.v, (short)0, c3, false, false);

        if (ks < 7) af = afn;
    }

    int rb = mtile * 16 + (lane >> 4) * 8;        // C: lanes 0-15 -> M=0..7, 16-31 -> M=8..15
    int colbase = ng * 64 + (lane & 15);
    v8f acc[4] = {c0, c1, c2, c3};
#pragma unroll
    for (int t = 0; t < 4; ++t) {
        int col = colbase + t * 16;
        float bvv = bias ? bias[col] : 0.0f;
#pragma unroll
        for (int r = 0; r < 8; ++r) {
            float v = acc[t][r] + bvv;
            size_t idx = (size_t)(rb + r) * 256 + col;
            if (BF16_OUT) ((unsigned short*)Cout)[idx] = f32_to_bf16_rne(v);
            else          ((float*)Cout)[idx] = v;
        }
    }
}

// ---------- 3. offsets/attn projection + softmax + sampling grid ----------
// One block per (b,q); 288 threads: t<192 -> offset cols, t>=192 -> attn logits.
__global__ __launch_bounds__(288)
void proj_softmax_kernel(const float* __restrict__ hidden,
                         const float* __restrict__ refpts,
                         const float* __restrict__ offW, const float* __restrict__ offB,
                         const float* __restrict__ attW, const float* __restrict__ attB,
                         float* __restrict__ grid, float* __restrict__ aw) {
    __shared__ float hs[256];
    __shared__ float offbuf[192];
    __shared__ float logit[96];
    int bq = blockIdx.x;                          // 0..4799
    int t  = threadIdx.x;                         // 0..287
    if (t < 256) hs[t] = hidden[(size_t)bq * 256 + t];
    __syncthreads();
    if (t < 192) {
        float acc = offB[t];
        for (int k = 0; k < 256; ++k) acc += hs[k] * offW[k * 192 + t];
        offbuf[t] = acc;
    } else {
        int cidx = t - 192;
        float acc = attB[cidx];
        for (int k = 0; k < 256; ++k) acc += hs[k] * attW[k * 96 + cidx];
        logit[cidx] = acc;
    }
    __syncthreads();
    if (t < 96) {                                 // t = h*12 + (l*4+p)
        int hh = t / 12, j = t % 12;
        const float* lg = &logit[hh * 12];
        float m = lg[0];
        for (int i = 1; i < 12; ++i) m = fmaxf(m, lg[i]);
        float s = 0.f;
        for (int i = 0; i < 12; ++i) s += __expf(lg[i] - m);
        aw[((size_t)bq * 8 + hh) * 12 + j] = __expf(lg[j] - m) / s;

        int l = j >> 2;
        const float* rp = refpts + ((size_t)bq * 3 + l) * 4;
        float lx = rp[0] + offbuf[t * 2 + 0] * 0.25f * rp[2] * 0.5f;
        float ly = rp[1] + offbuf[t * 2 + 1] * 0.25f * rp[3] * 0.5f;
        float* o = &grid[(((size_t)bq * 8 + hh) * 12 + j) * 2];
        o[0] = 2.f * lx - 1.f;
        o[1] = 2.f * ly - 1.f;
    }
}

// ---------- 4. bilinear sampling + weighted sum (value in bf16, L2-resident) ----
// One wave per (b,q,head); lane = channel (head_dim = 32).
__global__ __launch_bounds__(256)
void sample_kernel(const unsigned short* __restrict__ value16,
                   const float* __restrict__ grid,
                   const float* __restrict__ aw,
                   float* __restrict__ out) {
    int gwave = (int)((blockIdx.x * blockDim.x + threadIdx.x) >> 5); // (b*300+q)*8+h
    int lane  = threadIdx.x & 31;
    int h  = gwave & 7;
    int bq = gwave >> 3;
    int b  = bq / 300;
    const int HWdim[3]  = {80, 40, 20};
    const int Lstart[3] = {0, 6400, 8000};
    const float* gp = grid + (size_t)gwave * 24;
    const float* wp = aw   + (size_t)gwave * 12;
    float acc = 0.f;
#pragma unroll
    for (int l = 0; l < 3; ++l) {
        int Hh = HWdim[l], Ww = HWdim[l];
        size_t base = (size_t)b * 8400 + Lstart[l];
#pragma unroll
        for (int p = 0; p < 4; ++p) {
            int j = l * 4 + p;
            float x = (gp[j * 2 + 0] + 1.f) * 0.5f * Ww - 0.5f;
            float y = (gp[j * 2 + 1] + 1.f) * 0.5f * Hh - 0.5f;
            float x0 = floorf(x), y0 = floorf(y);
            float wx = x - x0, wy = y - y0;
            int x0i = (int)x0, y0i = (int)y0;
            float s = 0.f;
#pragma unroll
            for (int dy = 0; dy < 2; ++dy) {
#pragma unroll
                for (int dx = 0; dx < 2; ++dx) {
                    int xi = x0i + dx, yi = y0i + dy;
                    bool valid = (xi >= 0) & (xi < Ww) & (yi >= 0) & (yi < Hh);
                    int xc = min(max(xi, 0), Ww - 1);
                    int yc = min(max(yi, 0), Hh - 1);
                    float v = bf16_bits_to_f32(
                        value16[(base + (size_t)yc * Ww + xc) * 256 + h * 32 + lane]);
                    v = valid ? v : 0.f;
                    float cw = (dx ? wx : 1.f - wx) * (dy ? wy : 1.f - wy);
                    s += cw * v;
                }
            }
            acc += wp[j] * s;
        }
    }
    out[(size_t)bq * 256 + h * 32 + lane] = acc;  // channel order = h*32+d (matches ref)
}

// ---------- launch ----------
extern "C" void kernel_launch(void* const* d_in, const int* in_sizes, int n_in,
                              void* d_out, int out_size, void* d_ws, size_t ws_size,
                              hipStream_t stream) {
    const float* hidden   = (const float*)d_in[0];   // (16,300,256)
    const float* enc      = (const float*)d_in[1];   // (16,8400,256)
    const float* refpts   = (const float*)d_in[2];   // (16,300,3,4)
    const float* Wv       = (const float*)d_in[3];
    const float* bv       = (const float*)d_in[4];
    const float* Woff     = (const float*)d_in[5];
    const float* boff     = (const float*)d_in[6];
    const float* Watt     = (const float*)d_in[7];
    const float* batt     = (const float*)d_in[8];
    const float* Wout     = (const float*)d_in[9];
    const float* bout     = (const float*)d_in[10];

    char* ws = (char*)d_ws;
    size_t off = 0;
    unsigned short* value16 = (unsigned short*)(ws + off); off += (size_t)134400 * 256 * 2; // 68.8 MB (L2-resident)
    unsigned int*   WswV    = (unsigned int*)(ws + off);   off += 32768 * 4;                // 128 KB
    unsigned int*   WswO    = (unsigned int*)(ws + off);   off += 32768 * 4;                // 128 KB
    float*          grid    = (float*)(ws + off);          off += (size_t)4800 * 8 * 12 * 2 * 4;
    float*          awbuf   = (float*)(ws + off);          off += (size_t)4800 * 8 * 12 * 4;
    float*          attnout = (float*)(ws + off);          off += (size_t)4800 * 256 * 4;
    (void)ws_size; (void)in_sizes; (void)n_in; (void)out_size;

    // 1. swizzle both weight matrices into bf16 B-fragment layout
    swizzle_w_kernel<<<128, 256, 0, stream>>>(Wv,   WswV);
    swizzle_w_kernel<<<128, 256, 0, stream>>>(Wout, WswO);

    // 2. value projection: (134400 x 256) x (256 x 256) -> bf16 value
    //    grid = (8400/8 m-groups) * 4 n-groups
    gemm_bf16_wmma_kernel<true><<<(8400 / 8) * 4, 256, 0, stream>>>(enc, WswV, bv, value16, 8400);

    // 3. offsets/attn projection + softmax + grids
    proj_softmax_kernel<<<4800, 288, 0, stream>>>(hidden, refpts, Woff, boff, Watt, batt,
                                                  grid, awbuf);

    // 4. deformable sampling: 38400 waves, 8 waves/block
    sample_kernel<<<4800, 256, 0, stream>>>(value16, grid, awbuf, attnout);

    // 5. output projection: (4800 x 256) x (256 x 256) -> f32 d_out
    //    300 m-tiles -> 38 m-groups (last partially filled) * 4 n-groups
    gemm_bf16_wmma_kernel<false><<<38 * 4, 256, 0, stream>>>(attnout, WswO, bout,
                                                             d_out, 300);
}
